// FlashAttention_24292335027122
// MI455X (gfx1250) — compile-verified, matrix operand order checked by reading, not run on hardware
//
#include <hip/hip_runtime.h>
#include <hip/hip_bf16.h>

// ---------------------------------------------------------------------------
// MHA block for MI455X (gfx1250, wave32, WMMA).
//   B=2, S=2048, D=1024, H=16, HD=64.  ~69 GFLOP vs ~100MB -> compute bound.
//   All matmuls via v_wmma_f32_16x16x32_bf16 (fp32 accumulate).
//   Attention computed transposed (S^T = K Q^T, O^T = V^T P^T) so the softmax
//   is per-lane and P re-packs into a WMMA fragment in registers.
//   K/V tiles are staged block-wide into LDS with the CDNA5 async engine
//   (global_load_async_to_lds_b128 + s_wait_asynccnt), 4-deep LDS ring.
// ---------------------------------------------------------------------------

typedef unsigned short u16;
typedef unsigned int   u32;
typedef __attribute__((ext_vector_type(16))) __bf16 v16bf;
typedef __attribute__((ext_vector_type(8)))  float  v8f;

#define BB 2
#define SS 2048
#define DD 1024
#define HH 16
#define HD 64
#define NTOK (BB * SS)          // 4096

__device__ __forceinline__ u16 f2bf(float f) {
    union { float f; u32 u; } x; x.f = f;
    u32 r = x.u + 0x7FFFu + ((x.u >> 16) & 1u);   // round-to-nearest-even
    return (u16)(r >> 16);
}

// A/B fragment load for v_wmma_f32_16x16x32_bf16 from a row-major operand
// (rows = M for A, rows = N for B^T-style operands; K contiguous).
// Per 7.12.2: lane holds row = lane%16; elements are K chunks
// {h8..h8+7} and {16+h8..23+h8}, h8 = 8*(lane/16).  -> two b128 loads.
// Works for global pointers and for LDS-derived generic pointers (ds_load_b128).
__device__ __forceinline__ v16bf load_frag(const u16* base, int ld, int k0) {
    int lane = threadIdx.x & 31;
    const u16* p = base + (lane & 15) * ld + k0 + ((lane >> 4) << 3);
    union { uint4 u[2]; v16bf v; } f;
    f.u[0] = *(const uint4*)(p);
    f.u[1] = *(const uint4*)(p + 16);
    return f.v;
}

__device__ __forceinline__ v8f wmma_bf16(v16bf a, v16bf b, v8f c) {
    return __builtin_amdgcn_wmma_f32_16x16x32_bf16(
        /*neg_a=*/false, a, /*neg_b=*/false, b,
        /*c_mod=*/(short)0, c, /*reuse_a=*/false, /*reuse_b=*/false);
}

// Async copy of 16 bytes/lane from global to LDS (ASYNCcnt-tracked, GV mode).
__device__ __forceinline__ void async_cp16(u32 lds_off, const u16* gsrc) {
    asm volatile("global_load_async_to_lds_b128 %0, %1, off"
                 :: "v"(lds_off), "v"(gsrc) : "memory");
}

// ---------------------------------------------------------------------------
// fp32 -> bf16 conversion, 4 elements / thread
// ---------------------------------------------------------------------------
__global__ void __launch_bounds__(256)
cvt_kernel(const float4* __restrict__ in, u16* __restrict__ out, int n4) {
    int i = blockIdx.x * blockDim.x + threadIdx.x;
    if (i < n4) {
        float4 f = in[i];
        uint2 o;
        o.x = (u32)f2bf(f.x) | ((u32)f2bf(f.y) << 16);
        o.y = (u32)f2bf(f.z) | ((u32)f2bf(f.w) << 16);
        *(uint2*)(out + i * 4) = o;
    }
}

// ---------------------------------------------------------------------------
// Generic GEMM: Y[M=4096, N=1024] = A(bf16, MxK row-major) @ W(bf16, NxK)^T + b
// Each wave: one 16x64 output tile (1 M-tile x 4 N-tiles), K-loop of 32.
// MODE 0: -> Q head-layout [(b*16+h)*2048+s][64], scaled by 1/sqrt(HD)=0.125
// MODE 1: -> K head-layout
// MODE 2: -> V transposed  [(b*16+h)*64+d][2048]
// MODE 3: -> fp32 token-layout [tok][1024]  (final projection, into d_out)
// ---------------------------------------------------------------------------
template <int MODE>
__global__ void __launch_bounds__(256)
gemm_kernel(const u16* __restrict__ A, const u16* __restrict__ W,
            const float* __restrict__ bias, u16* __restrict__ obf,
            float* __restrict__ of32) {
    int wave = blockIdx.x * 8 + (threadIdx.x >> 5);
    int m0 = (wave >> 4) * 16;          // 256 M-tiles
    int n0 = (wave & 15) * 64;          // 16 groups of 4 N-tiles

    v8f acc[4] = {};
    const u16* arow = A + m0 * DD;
    #pragma unroll 4
    for (int k0 = 0; k0 < DD; k0 += 32) {
        v16bf af = load_frag(arow, DD, k0);
        #pragma unroll
        for (int t = 0; t < 4; ++t) {
            v16bf bf_ = load_frag(W + (n0 + t * 16) * DD, DD, k0);
            acc[t] = wmma_bf16(af, bf_, acc[t]);
        }
    }

    int lane = threadIdx.x & 31;
    int half = lane >> 4;
    int col  = lane & 15;
    #pragma unroll
    for (int t = 0; t < 4; ++t) {
        int n = n0 + t * 16 + col;
        float bv = bias[n];
        #pragma unroll
        for (int r = 0; r < 8; ++r) {
            int m = m0 + r + 8 * half;                 // token index
            float val = acc[t][r] + bv;
            if (MODE == 0) {
                int b = m >> 11, s = m & (SS - 1), h = n >> 6, d = n & (HD - 1);
                obf[(((b << 4) | h) * SS + s) * HD + d] = f2bf(val * 0.125f);
            } else if (MODE == 1) {
                int b = m >> 11, s = m & (SS - 1), h = n >> 6, d = n & (HD - 1);
                obf[(((b << 4) | h) * SS + s) * HD + d] = f2bf(val);
            } else if (MODE == 2) {
                int b = m >> 11, s = m & (SS - 1), h = n >> 6, d = n & (HD - 1);
                obf[(((b << 4) | h) * HD + d) * SS + s] = f2bf(val);
            } else {
                of32[m * DD + n] = val;
            }
        }
    }
}

// ---------------------------------------------------------------------------
// Flash attention.  One block = 8 waves = 8 query tiles of ONE (b,h) head,
// so K/V tiles are staged once per block into a 4-deep LDS ring via the
// async engine; every wave then feeds its WMMAs with ds_load_b128.
//   S^T[key,q] = K · Q^T   (A = K rows, B = Q rows;  C: slots=key, lanes=q)
//   O^T[d,q]  += V^T · P^T (A = V^T rows, B = P^T;   C: slots=d,   lanes=q)
// Each lane owns one query: 15 in-lane max/sum ops + one shfl_xor(16); the
// exp'd P slots are already in B-fragment element order (register re-pack).
// ---------------------------------------------------------------------------
__global__ void __launch_bounds__(256)
attn_kernel(const u16* __restrict__ Qh, const u16* __restrict__ Kh,
            const u16* __restrict__ Vt, u16* __restrict__ AO) {
    // Ring of 4 (K,V) tile pairs: K tile 32x64 bf16, V^T tile 64x32 bf16.
    __shared__ __align__(16) u16 Kl[4][32 * HD];   // 4 x 4KB
    __shared__ __align__(16) u16 Vl[4][HD * 32];   // 4 x 4KB

    int tid  = threadIdx.x;
    int lane = tid & 31;
    int bh   = blockIdx.x >> 4;                    // all 8 waves: same head
    int mt   = (blockIdx.x & 15) * 8 + (tid >> 5); // query tile 0..127

    const u16* kg = Kh + bh * SS * HD;             // head-local K  [S  x HD]
    const u16* vg = Vt + bh * HD * SS;             // head-local V^T[HD x S ]

    // Per-thread staging assignment (16B each): K: row=tid/8, chunk=tid%8;
    // V: row=tid/4, chunk=tid%4.  One K + one V async op per thread per tile.
    int krow = tid >> 3, kcol = (tid & 7) * 8;
    int vrow = tid >> 2, vcol = (tid & 3) * 8;

    // Q as B fragments (lane = query, elements = d-chunks), loaded once.
    const u16* qbase = Qh + (bh * SS + mt * 16) * HD;
    v16bf bq0 = load_frag(qbase, HD, 0);
    v16bf bq1 = load_frag(qbase, HD, 32);

    v8f o[4] = {};                       // O^T: frag t holds d = 16t + slot
    float mrun = -1e30f, lrun = 0.0f;    // per-lane (per-query) stats

    const int NKT = SS / 32;             // 64 key tiles
    // Prologue: stage tiles 0 and 1.
    #pragma unroll
    for (int pk = 0; pk < 2; ++pk) {
        async_cp16((u32)(size_t)&Kl[pk][krow * HD + kcol],
                   kg + (pk * 32 + krow) * HD + kcol);
        async_cp16((u32)(size_t)&Vl[pk][vrow * 32 + vcol],
                   vg + vrow * SS + pk * 32 + vcol);
    }

    for (int kt = 0; kt < NKT; ++kt) {
        // Oldest outstanding stage (tile kt: 2 ops) must retire; tile kt+1
        // may stay in flight.  Async loads complete in order.
        asm volatile("s_wait_asynccnt 0x2" ::: "memory");
        __syncthreads();                 // stage(kt) visible to all waves;
                                         // also: all waves done reading the
                                         // ring slot that stage(kt+2) reuses.
        {   // Stage tile kt+2 into ring slot (kt+2)&3 (clamped, uniform).
            int nt = kt + 2 < NKT ? kt + 2 : NKT - 1;
            int nb = (kt + 2) & 3;
            async_cp16((u32)(size_t)&Kl[nb][krow * HD + kcol],
                       kg + (nt * 32 + krow) * HD + kcol);
            async_cp16((u32)(size_t)&Vl[nb][vrow * 32 + vcol],
                       vg + vrow * SS + nt * 32 + vcol);
        }

        int buf = kt & 3;
        const u16* kb = &Kl[buf][0];

        // S^T tile: c0 = keys +0..15, c1 = keys +16..31  (ds_load_b128 feeds)
        v8f c0 = {}, c1 = {};
        c0 = wmma_bf16(load_frag(kb, HD, 0),           bq0, c0);
        c0 = wmma_bf16(load_frag(kb, HD, 32),          bq1, c0);
        c1 = wmma_bf16(load_frag(kb + 16 * HD, HD, 0), bq0, c1);
        c1 = wmma_bf16(load_frag(kb + 16 * HD, HD, 32),bq1, c1);

        // ---- online softmax, per lane ----
        float vmax = fmaxf(c0[0], c1[0]);
        #pragma unroll
        for (int r = 1; r < 8; ++r) vmax = fmaxf(vmax, fmaxf(c0[r], c1[r]));
        vmax = fmaxf(vmax, __shfl_xor(vmax, 16));      // partner lane combine
        float mnew  = fmaxf(mrun, vmax);
        float scale = __expf(mrun - mnew);

        union { u16 s[16]; v16bf v; } pf;              // P^T as B fragment
        float ts = 0.0f;
        #pragma unroll
        for (int r = 0; r < 8; ++r) {
            float e0 = __expf(c0[r] - mnew);           // key 8h + r
            float e1 = __expf(c1[r] - mnew);           // key 16 + 8h + r
            ts += e0 + e1;
            pf.s[r]     = f2bf(e0);                    // chunk0 element r
            pf.s[r + 8] = f2bf(e1);                    // chunk1 element r
        }
        ts += __shfl_xor(ts, 16);
        lrun = lrun * scale + ts;
        mrun = mnew;

        #pragma unroll
        for (int t = 0; t < 4; ++t)
            #pragma unroll
            for (int r = 0; r < 8; ++r) o[t][r] *= scale;

        // O^T += V^T_tile @ P^T  (V^T rows contiguous along keys in LDS)
        #pragma unroll
        for (int t = 0; t < 4; ++t)
            o[t] = wmma_bf16(load_frag(&Vl[buf][(t * 16) * 32], 32, 0), pf.v, o[t]);
    }

    // Normalize (per-lane) and emit bf16 attention output, token layout.
    float inv = 1.0f / lrun;
    int q = lane & 15, half = lane >> 4;
    int b = bh >> 4, h = bh & 15;
    u16* ob = AO + (b * SS + mt * 16 + q) * DD + h * HD;
    #pragma unroll
    for (int t = 0; t < 4; ++t)
        #pragma unroll
        for (int r = 0; r < 8; ++r)
            ob[t * 16 + r + 8 * half] = f2bf(o[t][r] * inv);
}

// ---------------------------------------------------------------------------
extern "C" void kernel_launch(void* const* d_in, const int* in_sizes, int n_in,
                              void* d_out, int out_size, void* d_ws, size_t ws_size,
                              hipStream_t stream) {
    (void)in_sizes; (void)n_in; (void)out_size; (void)ws_size;

    const float* x  = (const float*)d_in[0];
    const float* Wq = (const float*)d_in[1];
    const float* bq = (const float*)d_in[2];
    const float* Wk = (const float*)d_in[3];
    const float* bk = (const float*)d_in[4];
    const float* Wv = (const float*)d_in[5];
    const float* bv = (const float*)d_in[6];
    const float* Wo = (const float*)d_in[7];
    const float* bo = (const float*)d_in[8];
    float* out = (float*)d_out;

    const int XN = NTOK * DD;     // 4M elems
    const int WN = DD * DD;       // 1M elems

    u16* ws  = (u16*)d_ws;        // ~50.3 MB of bf16 staging
    u16* Xbf = ws;
    u16* Wqb = Xbf + XN;
    u16* Wkb = Wqb + WN;
    u16* Wvb = Wkb + WN;
    u16* Wob = Wvb + WN;
    u16* Qh  = Wob + WN;
    u16* Kh  = Qh + XN;
    u16* Vt  = Kh + XN;
    u16* AOb = Vt + XN;

    // 1) bf16 staging
    cvt_kernel<<<XN / 4 / 256, 256, 0, stream>>>((const float4*)x,  Xbf, XN / 4);
    cvt_kernel<<<WN / 4 / 256, 256, 0, stream>>>((const float4*)Wq, Wqb, WN / 4);
    cvt_kernel<<<WN / 4 / 256, 256, 0, stream>>>((const float4*)Wk, Wkb, WN / 4);
    cvt_kernel<<<WN / 4 / 256, 256, 0, stream>>>((const float4*)Wv, Wvb, WN / 4);
    cvt_kernel<<<WN / 4 / 256, 256, 0, stream>>>((const float4*)Wo, Wob, WN / 4);

    // 2) QKV projections (4096 wave-tiles each -> 512 blocks of 8 waves)
    gemm_kernel<0><<<512, 256, 0, stream>>>(Xbf, Wqb, bq, Qh, nullptr);
    gemm_kernel<1><<<512, 256, 0, stream>>>(Xbf, Wkb, bk, Kh, nullptr);
    gemm_kernel<2><<<512, 256, 0, stream>>>(Xbf, Wvb, bv, Vt, nullptr);

    // 3) flash attention: block = one head x 8 query tiles, async K/V staging
    attn_kernel<<<512, 256, 0, stream>>>(Qh, Kh, Vt, AOb);

    // 4) output projection -> fp32 d_out
    gemm_kernel<3><<<512, 256, 0, stream>>>(AOb, Wob, bo, nullptr, out);
}
